// vqvae1d_21998822490429
// MI455X (gfx1250) — matrix operand order repaired and checked by reading, not compiled, not run on hardware
//
#include <hip/hip_runtime.h>

// ---------------------------------------------------------------------------
// VQ-VAE-1D inference for MI455X (gfx1250): all conv / linear layers lowered
// to bf16 WMMA GEMMs (v_wmma_f32_16x16x32_bf16), activations channels-last.
// GEMM wave tile 64x32 (8 WMMA / K-step); named pointers so loads stay
// global_load_b128 (no flat fallback).
// ---------------------------------------------------------------------------

typedef unsigned short u16;
typedef unsigned int   u32;
typedef __attribute__((ext_vector_type(4)))  u32    u32x4;
typedef __attribute__((ext_vector_type(16))) __bf16 v16bf;
typedef __attribute__((ext_vector_type(8)))  float  v8f;

union Frag32B { u32x4 u[2]; v16bf v; };

__device__ __forceinline__ u16 f2bf(float f) {
  u32 u = __float_as_uint(f);
  u32 r = ((u >> 16) & 1u) + 0x7FFFu;   // round-to-nearest-even
  return (u16)((u + r) >> 16);
}

// ------------------------- weight pack (+BN fold) --------------------------
// src torch layout (Cout, Cin, K) fp32  ->  dst bf16 [Cout][K*Cin], kk=k*Cin+ci
__global__ __launch_bounds__(256)
void pack_w_kernel(const float* __restrict__ wt, const float* __restrict__ cbias,
                   const float* __restrict__ g, const float* __restrict__ bb,
                   const float* __restrict__ m, const float* __restrict__ v,
                   u16* __restrict__ Wp, float* __restrict__ effB,
                   int Cout, int Cin, int K)
{
  int i = blockIdx.x * blockDim.x + threadIdx.x;
  if (i < Cout) {
    float s = 1.f, sh = 0.f;
    if (g) { s = g[i] * rsqrtf(v[i] + 1e-5f); sh = bb[i] - m[i] * s; }
    effB[i] = cbias[i] * s + sh;
  }
  int total = Cout * Cin * K;
  if (i >= total) return;
  int co = i / (K * Cin);
  int r  = i % (K * Cin);
  int k  = r / Cin;
  int ci = r % Cin;
  float s = 1.f;
  if (g) s = g[co] * rsqrtf(v[co] + 1e-5f);
  Wp[i] = f2bf(wt[((long)co * Cin + ci) * K + k] * s);
}

// --------------------- x (B,C,T) fp32 -> [b*T+t][C] bf16 -------------------
__global__ __launch_bounds__(256)
void x_to_bf16_kernel(const float* __restrict__ x, u16* __restrict__ xb,
                      int B, int C, int T)
{
  long i = (long)blockIdx.x * blockDim.x + threadIdx.x;  // enumerates (b,t,c)
  if (i >= (long)B * C * T) return;
  int c = (int)(i % C);
  long rem = i / C;
  int t = (int)(rem % T);
  int b = (int)(rem / T);
  xb[i] = f2bf(x[((long)b * C + c) * T + t]);
}

// ----------------------------- im2col (bf16) -------------------------------
// act: [B*Tin][Cin] bf16 ; colT: [nchunk][4*Cin] bf16, kk = k*Cin + ci
// transposed=0: conv  K=4,s=2,p=1   ti = 2t-1+k
// transposed=1: convT K=4,s=2,p=1   ti = (t+1-k)/2 valid iff even & in-range
__global__ __launch_bounds__(256)
void im2col_kernel(const u16* __restrict__ act, u16* __restrict__ colT,
                   int Cin, int Tin, int Tout, int n0, int nchunk, int transposed)
{
  int KK = 4 * Cin;
  long total = (long)nchunk * KK / 8;
  long i = (long)blockIdx.x * blockDim.x + threadIdx.x;
  if (i >= total) return;
  long e  = i * 8;
  int nl  = (int)(e / KK);
  int kk  = (int)(e % KK);
  int k   = kk / Cin;
  int ci  = kk % Cin;
  int n   = n0 + nl;
  int b   = n / Tout;
  int t   = n % Tout;
  int ti; bool valid;
  if (!transposed) { ti = 2 * t - 1 + k; valid = (ti >= 0) && (ti < Tin); }
  else { int num = t + 1 - k; ti = num >> 1;
         valid = (num >= 0) && ((num & 1) == 0) && (ti < Tin); }
  u32x4 val = (u32x4)0;
  if (valid) val = *(const u32x4*)(act + ((long)b * Tin + ti) * Cin + ci);
  *(u32x4*)(colT + (long)nl * KK + kk) = val;
}

// --------------------------- WMMA bf16 GEMM --------------------------------
// D[M][N] = A[M][Kd] * Bt[N][Kd]^T + bias.
// 8 waves as 2(M)x4(N); wave tile 64(M)x32(N) = 8 accumulators.
// out16  : bf16 [N][M] (channels-last activation), nullable
// out32nd: fp32 [N][M], nullable
// outbct : fp32 scatter to (B, M, Tout), nullable (n global = n_glob0 + n)
__global__ __launch_bounds__(256)
void wmma_gemm_bf16(const u16* __restrict__ A, const u16* __restrict__ Bt,
                    const float* __restrict__ bias, int M, int Kd,
                    u16* __restrict__ out16, float* __restrict__ out32nd,
                    float* __restrict__ outbct, int n_glob0, int T_out, int CT)
{
  int lane = threadIdx.x & 31;
  int w    = threadIdx.x >> 5;
  int m0   = blockIdx.x * 128 + (w >> 2) * 64;
  int n0   = blockIdx.y * 128 + (w & 3)  * 32;
  if (m0 + 64 > M) return;                 // wave-uniform guard (M % 64 == 0)
  int lm = lane & 15;
  int lg = lane >> 4;

  v8f c00 = {}, c01 = {}, c10 = {}, c11 = {};
  v8f c20 = {}, c21 = {}, c30 = {}, c31 = {};

  const u16* a0p = A  + (long)(m0 +  0 + lm) * Kd + lg * 8;
  const u16* a1p = A  + (long)(m0 + 16 + lm) * Kd + lg * 8;
  const u16* a2p = A  + (long)(m0 + 32 + lm) * Kd + lg * 8;
  const u16* a3p = A  + (long)(m0 + 48 + lm) * Kd + lg * 8;
  const u16* b0p = Bt + (long)(n0 +  0 + lm) * Kd + lg * 8;
  const u16* b1p = Bt + (long)(n0 + 16 + lm) * Kd + lg * 8;

  for (int k0 = 0; k0 < Kd; k0 += 32) {
    Frag32B a0, a1, a2, a3, b0, b1;
    a0.u[0] = *(const u32x4*)(a0p); a0.u[1] = *(const u32x4*)(a0p + 16);
    a1.u[0] = *(const u32x4*)(a1p); a1.u[1] = *(const u32x4*)(a1p + 16);
    a2.u[0] = *(const u32x4*)(a2p); a2.u[1] = *(const u32x4*)(a2p + 16);
    a3.u[0] = *(const u32x4*)(a3p); a3.u[1] = *(const u32x4*)(a3p + 16);
    b0.u[0] = *(const u32x4*)(b0p); b0.u[1] = *(const u32x4*)(b0p + 16);
    b1.u[0] = *(const u32x4*)(b1p); b1.u[1] = *(const u32x4*)(b1p + 16);
    c00 = __builtin_amdgcn_wmma_f32_16x16x32_bf16(false, a0.v, false, b0.v, (short)0, c00, false, false);
    c10 = __builtin_amdgcn_wmma_f32_16x16x32_bf16(false, a1.v, false, b0.v, (short)0, c10, false, false);
    c20 = __builtin_amdgcn_wmma_f32_16x16x32_bf16(false, a2.v, false, b0.v, (short)0, c20, false, false);
    c30 = __builtin_amdgcn_wmma_f32_16x16x32_bf16(false, a3.v, false, b0.v, (short)0, c30, false, false);
    c01 = __builtin_amdgcn_wmma_f32_16x16x32_bf16(false, a0.v, false, b1.v, (short)0, c01, false, false);
    c11 = __builtin_amdgcn_wmma_f32_16x16x32_bf16(false, a1.v, false, b1.v, (short)0, c11, false, false);
    c21 = __builtin_amdgcn_wmma_f32_16x16x32_bf16(false, a2.v, false, b1.v, (short)0, c21, false, false);
    c31 = __builtin_amdgcn_wmma_f32_16x16x32_bf16(false, a3.v, false, b1.v, (short)0, c31, false, false);
    a0p += 32; a1p += 32; a2p += 32; a3p += 32;
    b0p += 32; b1p += 32;
  }

  auto emit = [&](v8f cc, int mb, int nb) {
    int mr = mb + lg * 8;                  // 8 consecutive M rows
    int n  = nb + lm;
    float vals[8];
#pragma unroll
    for (int r = 0; r < 8; ++r) vals[r] = cc[r] + bias[mr + r];
    if (out16) {
      u32x4 pk;
      pk.x = (u32)f2bf(vals[0]) | ((u32)f2bf(vals[1]) << 16);
      pk.y = (u32)f2bf(vals[2]) | ((u32)f2bf(vals[3]) << 16);
      pk.z = (u32)f2bf(vals[4]) | ((u32)f2bf(vals[5]) << 16);
      pk.w = (u32)f2bf(vals[6]) | ((u32)f2bf(vals[7]) << 16);
      *(u32x4*)(out16 + (long)n * M + mr) = pk;
    }
    if (out32nd) {
#pragma unroll
      for (int r = 0; r < 8; ++r) out32nd[(long)n * M + mr + r] = vals[r];
    }
    if (outbct) {
      int ng = n_glob0 + n;
      int b  = ng / T_out, t = ng % T_out;
      long base = (long)b * CT + t;
#pragma unroll
      for (int r = 0; r < 8; ++r) outbct[base + (long)(mr + r) * T_out] = vals[r];
    }
  };
  emit(c00, m0 +  0, n0 +  0);
  emit(c10, m0 + 16, n0 +  0);
  emit(c20, m0 + 32, n0 +  0);
  emit(c30, m0 + 48, n0 +  0);
  emit(c01, m0 +  0, n0 + 16);
  emit(c11, m0 + 16, n0 + 16);
  emit(c21, m0 + 32, n0 + 16);
  emit(c31, m0 + 48, n0 + 16);
}

// ------------------------------ VQ + loss ----------------------------------
#define NCODES 50
#define EDIM   192
__global__ void init_loss_kernel(float* p) { if (threadIdx.x == 0) *p = 0.f; }

__global__ __launch_bounds__(256)
void vq_kernel(const float* __restrict__ z, const float* __restrict__ cb,
               u16* __restrict__ qt, float* __restrict__ loss, int N)
{
  __shared__ float scb[NCODES * EDIM];
  for (int i = threadIdx.x; i < NCODES * EDIM; i += blockDim.x) scb[i] = cb[i];
  __syncthreads();
  int n = blockIdx.x * blockDim.x + threadIdx.x;
  if (n >= N) return;
  const float* zr = z + (long)n * EDIM;
  float acc[NCODES];
#pragma unroll
  for (int k = 0; k < NCODES; ++k) acc[k] = 0.f;
  for (int d = 0; d < EDIM; ++d) {
    float zv = zr[d];
#pragma unroll
    for (int k = 0; k < NCODES; ++k) {
      float df = zv - scb[k * EDIM + d];
      acc[k] += df * df;
    }
  }
  float best = acc[0]; int bi = 0;
#pragma unroll
  for (int k = 1; k < NCODES; ++k) if (acc[k] < best) { best = acc[k]; bi = k; }
  for (int d = 0; d < EDIM; ++d) qt[(long)n * EDIM + d] = f2bf(scb[bi * EDIM + d]);
  atomicAdd(loss, best * (1.0f / ((float)N * (float)EDIM)));  // COMMIT = 1.0
}

// ------------------------------- host side ---------------------------------
static inline int divup(int a, int b) { return (a + b - 1) / b; }

extern "C" void kernel_launch(void* const* d_in, const int* in_sizes, int n_in,
                              void* d_out, int out_size, void* d_ws, size_t ws_size,
                              hipStream_t stream)
{
  (void)in_sizes; (void)n_in; (void)out_size; (void)ws_size;
  // setup_inputs() dict order: x, enc_w[3], enc_b[3], enc_bn_g[2], enc_bn_b[2],
  // enc_bn_m[2], enc_bn_v[2], lxm_w, lxm_b, codebook, dec_embed_w, dec_embed_b,
  // dec_w[3], dec_b[3], dec_bn_g[2], dec_bn_b[2], dec_bn_m[2], dec_bn_v[2]
  const float* X    = (const float*)d_in[0];
  const float* EW0  = (const float*)d_in[1];
  const float* EW1  = (const float*)d_in[2];
  const float* EW2  = (const float*)d_in[3];
  const float* EB0  = (const float*)d_in[4];
  const float* EB1  = (const float*)d_in[5];
  const float* EB2  = (const float*)d_in[6];
  const float* EG0  = (const float*)d_in[7];
  const float* EG1  = (const float*)d_in[8];
  const float* EBB0 = (const float*)d_in[9];
  const float* EBB1 = (const float*)d_in[10];
  const float* EM0  = (const float*)d_in[11];
  const float* EM1  = (const float*)d_in[12];
  const float* EV0  = (const float*)d_in[13];
  const float* EV1  = (const float*)d_in[14];
  const float* LXW  = (const float*)d_in[15];
  const float* LXB  = (const float*)d_in[16];
  const float* CB   = (const float*)d_in[17];
  const float* DEW  = (const float*)d_in[18];
  const float* DEB  = (const float*)d_in[19];
  const float* DW0  = (const float*)d_in[20];
  const float* DW1  = (const float*)d_in[21];
  const float* DW2  = (const float*)d_in[22];
  const float* DB0  = (const float*)d_in[23];
  const float* DB1  = (const float*)d_in[24];
  const float* DB2  = (const float*)d_in[25];
  const float* DG0  = (const float*)d_in[26];
  const float* DG1  = (const float*)d_in[27];
  const float* DBB0 = (const float*)d_in[28];
  const float* DBB1 = (const float*)d_in[29];
  const float* DM0  = (const float*)d_in[30];
  const float* DM1  = (const float*)d_in[31];
  const float* DV0  = (const float*)d_in[32];
  const float* DV1  = (const float*)d_in[33];

  const int Bsz = 64;
  const int NCHUNK = 8192;

  char* wsp = (char*)d_ws;
  auto carve = [&](size_t bytes) -> void* {
    char* p = wsp; wsp += (bytes + 255) & ~(size_t)255; return (void*)p;
  };
  u16*   xb   = (u16*)  carve((size_t)Bsz * 1024 * 128 * 2);     // [b*T+t][128]
  u16*   actA = (u16*)  carve((size_t)8388608 * 2);              // 16 MB ping
  u16*   actB = (u16*)  carve((size_t)8388608 * 2);              // 16 MB pong
  u16*   colT = (u16*)  carve((size_t)2048 * NCHUNK * 2);        // 32 MB im2col
  float* z32  = (float*)carve((size_t)8192 * 192 * 4);
  u16*   qt   = (u16*)  carve((size_t)8192 * 192 * 2);
  u16* Wp1 = (u16*)carve((size_t)256 * 512  * 2);  float* eb1 = (float*)carve(256 * 4);
  u16* Wp2 = (u16*)carve((size_t)512 * 1024 * 2);  float* eb2 = (float*)carve(512 * 4);
  u16* Wp3 = (u16*)carve((size_t)512 * 2048 * 2);  float* eb3 = (float*)carve(512 * 4);
  u16* Wp4 = (u16*)carve((size_t)192 * 512  * 2);  float* eb4 = (float*)carve(192 * 4);
  u16* Wp5 = (u16*)carve((size_t)512 * 192  * 2);  float* eb5 = (float*)carve(512 * 4);
  u16* Wp6 = (u16*)carve((size_t)512 * 2048 * 2);  float* eb6 = (float*)carve(512 * 4);
  u16* Wp7 = (u16*)carve((size_t)256 * 2048 * 2);  float* eb7 = (float*)carve(256 * 4);
  u16* Wp8 = (u16*)carve((size_t)128 * 1024 * 2);  float* eb8 = (float*)carve(128 * 4);

  auto pack = [&](const float* wt, const float* cb2, const float* g, const float* bb,
                  const float* m, const float* v, u16* Wp, float* eb,
                  int Co, int Ci, int K) {
    int total = Co * Ci * K;
    pack_w_kernel<<<divup(total, 256), 256, 0, stream>>>(wt, cb2, g, bb, m, v, Wp, eb, Co, Ci, K);
  };
  pack(EW0, EB0, EG0, EBB0, EM0, EV0, Wp1, eb1, 256, 128, 4);
  pack(EW1, EB1, EG1, EBB1, EM1, EV1, Wp2, eb2, 512, 256, 4);
  pack(EW2, EB2, nullptr, nullptr, nullptr, nullptr, Wp3, eb3, 512, 512, 4);
  pack(LXW, LXB, nullptr, nullptr, nullptr, nullptr, Wp4, eb4, 192, 512, 1);
  pack(DEW, DEB, nullptr, nullptr, nullptr, nullptr, Wp5, eb5, 512, 192, 1);
  pack(DW0, DB0, DG0, DBB0, DM0, DV0, Wp6, eb6, 512, 512, 4);
  pack(DW1, DB1, DG1, DBB1, DM1, DV1, Wp7, eb7, 256, 512, 4);
  pack(DW2, DB2, nullptr, nullptr, nullptr, nullptr, Wp8, eb8, 128, 256, 4);

  x_to_bf16_kernel<<<divup(Bsz * 128 * 1024, 256), 256, 0, stream>>>(X, xb, Bsz, 128, 1024);

  auto conv_layer = [&](const u16* in_act, int Cin, int Tin, int Tout, int transposed,
                        const u16* Wp, const float* eb, int Cout,
                        u16* o16, float* obct) {
    int Ntot = Bsz * Tout;
    int KK = 4 * Cin;
    for (int n0 = 0; n0 < Ntot; n0 += NCHUNK) {
      long telems = (long)NCHUNK * KK / 8;
      im2col_kernel<<<divup((int)telems, 256), 256, 0, stream>>>(
          in_act, colT, Cin, Tin, Tout, n0, NCHUNK, transposed);
      dim3 g(divup(Cout, 128), NCHUNK / 128);
      wmma_gemm_bf16<<<g, 256, 0, stream>>>(
          Wp, colT, eb, Cout, KK,
          o16 ? o16 + (long)n0 * Cout : nullptr,
          nullptr, obct, n0, Tout, Cout * Tout);
    }
  };
  auto dense = [&](const u16* Wp, const float* eb, const u16* Bt,
                   int M, int Kd, int N, u16* o16, float* o32) {
    dim3 g(divup(M, 128), N / 128);
    wmma_gemm_bf16<<<g, 256, 0, stream>>>(Wp, Bt, eb, M, Kd, o16, o32,
                                          nullptr, 0, 1, 0);
  };

  // ---- encoder ----
  conv_layer(xb,   128, 1024, 512, 0, Wp1, eb1, 256, actA, nullptr);
  conv_layer(actA, 256,  512, 256, 0, Wp2, eb2, 512, actB, nullptr);
  conv_layer(actB, 512,  256, 128, 0, Wp3, eb3, 512, actA, nullptr);
  // lxm_embed: z32 = [8192][192] fp32
  dense(Wp4, eb4, actA, 192, 512, 8192, nullptr, z32);

  // ---- VQ + commitment loss ----
  float* outf = (float*)d_out;
  init_loss_kernel<<<1, 64, 0, stream>>>(outf);
  vq_kernel<<<8192 / 256, 256, 0, stream>>>(z32, CB, qt, outf, 8192);

  // ---- decoder ----
  dense(Wp5, eb5, qt, 512, 192, 8192, actB, nullptr);
  conv_layer(actB, 512, 128,  256, 1, Wp6, eb6, 512, actA, nullptr);
  conv_layer(actA, 512, 256,  512, 1, Wp7, eb7, 256, actB, nullptr);
  conv_layer(actB, 256, 512, 1024, 1, Wp8, eb8, 128, nullptr, outf + 1);
}
